// EGNNLayer_58471684768170
// MI455X (gfx1250) — matrix operand order repaired and checked by reading, not compiled, not run on hardware
//
#include <hip/hip_runtime.h>

typedef __attribute__((ext_vector_type(16))) _Float16 v16h;
typedef __attribute__((ext_vector_type(8)))  _Float16 v8h;
typedef __attribute__((ext_vector_type(4)))  _Float16 v4h;
typedef __attribute__((ext_vector_type(8)))  float    v8f;
typedef __attribute__((ext_vector_type(4)))  unsigned int u32x4;
typedef __attribute__((ext_vector_type(8)))  unsigned int u32x8;

#define HIDDEN 128
#define MDIM   128
#define NN     50000
#define NE     600000

// ---------------------------------------------------------------- helpers
__device__ __forceinline__ float sigmoidf_(float x) { return 1.0f / (1.0f + __expf(-x)); }
__device__ __forceinline__ float siluf_(float x)    { return x * sigmoidf_(x); }

__device__ __forceinline__ v8f wmma_f16(v16h a, v16h b, v8f c) {
    return __builtin_amdgcn_wmma_f32_16x16x32_f16(false, a, false, b, (short)0, c, false, false);
}

// A-fragment (16x32, f16) from LDS, per ISA 16-bit A layout:
// lanes 0-15 hold row M=lane, K = kb+0..7 then kb+16..23
// lanes 16-31 hold row M=lane-16, K = kb+8..15 then kb+24..31
__device__ __forceinline__ v16h load_a_frag(const _Float16* sm, int stride, int row_base, int kb) {
    const int lane = threadIdx.x & 31;
    const int m  = lane & 15;
    const int hi = lane >> 4;
    const _Float16* r = sm + (row_base + m) * stride + kb;
    v8h lo  = *(const v8h*)(r + (hi ? 8  : 0));
    v8h hi8 = *(const v8h*)(r + (hi ? 24 : 16));
    return __builtin_shufflevector(lo, hi8, 0,1,2,3,4,5,6,7,8,9,10,11,12,13,14,15);
}

// Tensor Data Mover: DMA one contiguous region (nelem8 x 8 bytes) from global
// into LDS at byte offset lds_off. D# per CDNA5 ISA ch.8:
//   group0: count=1, lds_addr, global_addr[56:0], type=2
//   group1: data_size=8B, tensor_dim0 = tile_dim0 = stride0 = nelem8, dims1+ = 1
// Issued by one wave; completion via TENSORcnt.
__device__ __forceinline__ void tdm_load_row(unsigned lds_off, const void* gptr, unsigned nelem8) {
    unsigned long long ga = (unsigned long long)gptr;
    u32x4 g0;
    g0[0] = 1u;                                                  // count=1, no gather
    g0[1] = lds_off;                                             // lds_addr (bytes)
    g0[2] = (unsigned)(ga & 0xffffffffu);                        // global_addr lo
    g0[3] = (unsigned)((ga >> 32) & 0x01ffffffu) | (2u << 30);   // addr hi | type=2
    u32x8 g1;
    g1[0] = 3u << 16;                                            // data_size = 8 bytes
    g1[1] = (nelem8 & 0xffffu) << 16;                            // tensor_dim0 lo16
    g1[2] = (nelem8 >> 16) | (1u << 16);                         // dim0 hi16 | tensor_dim1=1
    g1[3] = (nelem8 & 0xffffu) << 16;                            // tile_dim0
    g1[4] = 1u;                                                  // tile_dim1=1, tile_dim2=0
    g1[5] = nelem8;                                              // tensor_dim0_stride lo32
    g1[6] = (nelem8 & 0xffffu) << 16;                            // stride0 hi=0 | stride1 lo16
    g1[7] = 0u;                                                  // stride1 hi
    asm volatile("tensor_load_to_lds %0, %1" :: "s"(g0), "s"(g1) : "memory");
}

// ---------------------------------------------------------------- weight packing
// B-fragment layout (32x16, f16): lane n = nt*16 + (lane&15);
// lanes 0-15 hold K = kt*32+0..7,16..23, lanes 16-31 hold K = kt*32+8..15,24..31
__global__ void pack_b_kernel(const float* __restrict__ W, _Float16* __restrict__ dst,
                              int K_actual, int ktiles) {
    int t = blockIdx.x * blockDim.x + threadIdx.x;
    int total = ktiles * 256;            // ktiles * 8 ntiles * 32 lanes
    if (t >= total) return;
    int lane = t & 31;
    int nt   = (t >> 5) & 7;
    int kt   = t >> 8;
    int n    = nt * 16 + (lane & 15);
    int kb   = kt * 32 + ((lane >> 4) ? 8 : 0);
    _Float16* d = dst + t * 16;
#pragma unroll
    for (int i = 0; i < 16; ++i) {
        int k = kb + (i < 8 ? i : i + 8);
        float w = (k < K_actual) ? W[k * 128 + n] : 0.0f;
        d[i] = (_Float16)w;
    }
}

// ---------------------------------------------------------------- init
__global__ void init_kernel(float* __restrict__ maggr, const float* __restrict__ x,
                            float* __restrict__ xout) {
    long i = (long)blockIdx.x * blockDim.x + threadIdx.x;
    long stride = (long)gridDim.x * blockDim.x;
    for (long j = i; j < (long)NN * 128; j += stride) maggr[j] = 0.0f;
    for (long j = i; j < (long)NN * 3;   j += stride) xout[j]  = x[j];
}

// ---------------------------------------------------------------- edge kernel
__global__ __launch_bounds__(128)
void egnn_edge_kernel(const float* __restrict__ h, const float* __restrict__ x,
                      const int* __restrict__ eidx,
                      const _Float16* __restrict__ We1f, const float* __restrict__ be1,
                      const _Float16* __restrict__ We2Wc1f, const float* __restrict__ be2,
                      const float* __restrict__ Wg,  const float* __restrict__ bg,
                      const float* __restrict__ bc1,
                      const float* __restrict__ Wc2, const float* __restrict__ bc2,
                      float* __restrict__ maggr, float* __restrict__ xout) {
    __shared__ _Float16 sm_msg[64 * 288];   // [64 edges][288 K] (256 h, 1 dist, pad)
    __shared__ _Float16 sm_m  [64 * 128];   // activations between layers
    __shared__ __align__(64) _Float16 sm_bw[32768];  // TDM-staged We2 (16K halves) + Wc1 (16K halves)
    __shared__ float    sm_cd [64 * 3];
    __shared__ float    sm_gate[64];

    const int tid  = threadIdx.x;
    const int wave = tid >> 5;
    const int lane = tid & 31;
    const int e0   = blockIdx.x * 64;

    // ---- kick off TDM stage of layer-2 / coord weights (overlaps gather + layer 1)
    if (wave == 0)
        tdm_load_row((unsigned)(unsigned long long)&sm_bw[0], We2Wc1f, 8192 /* x8B = 64KB */);

    // ---- gather h rows: idx = edge*2 + part; part0 = h_i = h[col], part1 = h_j = h[row]
    for (int idx = wave; idx < 128; idx += 4) {
        int e = idx >> 1, part = idx & 1;
        int edge = e0 + e;
        int node = part ? eidx[edge] : eidx[NE + edge];
        float4 v = ((const float4*)(h + (long)node * HIDDEN))[lane];
        v4h p;
        p[0] = (_Float16)v.x; p[1] = (_Float16)v.y;
        p[2] = (_Float16)v.z; p[3] = (_Float16)v.w;
        *(v4h*)&sm_msg[e * 288 + part * 128 + lane * 4] = p;
    }
    // ---- dist, coord_diff, zero-pad
    if (tid < 64) {
        int edge = e0 + tid;
        int r = eidx[edge], c = eidx[NE + edge];
        float dx = x[c * 3 + 0] - x[r * 3 + 0];
        float dy = x[c * 3 + 1] - x[r * 3 + 1];
        float dz = x[c * 3 + 2] - x[r * 3 + 2];
        sm_cd[tid * 3 + 0] = dx; sm_cd[tid * 3 + 1] = dy; sm_cd[tid * 3 + 2] = dz;
        float d = sqrtf(dx * dx + dy * dy + dz * dz);
        sm_msg[tid * 288 + 256] = (_Float16)d;
        for (int k = 257; k < 288; ++k) sm_msg[tid * 288 + k] = (_Float16)0.0f;
    }
    __syncthreads();

    const int m15 = lane & 15, hi = lane >> 4;
    const int row_base = wave * 16;
    v8f acc[8];

    // ---- layer 1: [64x288] @ [288x128]  (9 K-steps, 8 N-tiles), B from global
    const v16h* B1 = (const v16h*)We1f;
#pragma unroll
    for (int nt = 0; nt < 8; ++nt) {
        float b = be1[nt * 16 + m15];
        v8f c;
#pragma unroll
        for (int i = 0; i < 8; ++i) c[i] = b;
#pragma unroll
        for (int kt = 0; kt < 9; ++kt) {
            v16h a = load_a_frag(sm_msg, 288, row_base, kt * 32);
            c = wmma_f16(a, B1[(kt * 8 + nt) * 32 + lane], c);
        }
        acc[nt] = c;
    }
#pragma unroll
    for (int nt = 0; nt < 8; ++nt)
#pragma unroll
        for (int v = 0; v < 8; ++v)
            sm_m[(row_base + v + 8 * hi) * 128 + nt * 16 + m15] = (_Float16)siluf_(acc[nt][v]);

    if (wave == 0) __builtin_amdgcn_s_wait_tensorcnt(0);  // staged weights landed
    __syncthreads();

    // ---- layer 2: [64x128] @ [128x128], B from LDS (TDM-staged)
    const v16h* B2 = (const v16h*)sm_bw;
#pragma unroll
    for (int nt = 0; nt < 8; ++nt) {
        float b = be2[nt * 16 + m15];
        v8f c;
#pragma unroll
        for (int i = 0; i < 8; ++i) c[i] = b;
#pragma unroll
        for (int kt = 0; kt < 4; ++kt) {
            v16h a = load_a_frag(sm_m, 128, row_base, kt * 32);
            c = wmma_f16(a, B2[(kt * 8 + nt) * 32 + lane], c);
        }
        acc[nt] = c;
    }
    __syncthreads();   // all reads of m1 done before overwrite
#pragma unroll
    for (int nt = 0; nt < 8; ++nt)
#pragma unroll
        for (int v = 0; v < 8; ++v)
            sm_m[(row_base + v + 8 * hi) * 128 + nt * 16 + m15] = (_Float16)siluf_(acc[nt][v]);
    __syncthreads();

    // ---- gate = sigmoid(m2 @ Wg + bg) per edge (vectorized LDS reads)
    if (tid < 64) {
        float s = bg[0];
        const v8h* row = (const v8h*)&sm_m[tid * 128];
        for (int k8 = 0; k8 < 16; ++k8) {
            v8h v = row[k8];
#pragma unroll
            for (int i = 0; i < 8; ++i) s += (float)v[i] * Wg[k8 * 8 + i];
        }
        sm_gate[tid] = sigmoidf_(s);
    }
    __syncthreads();
    // ---- apply gate in place
    {
        int e = tid >> 1, kb = (tid & 1) * 64;
        float g = sm_gate[e];
        for (int k = 0; k < 64; ++k) {
            int i = e * 128 + kb + k;
            sm_m[i] = (_Float16)((float)sm_m[i] * g);
        }
    }
    __syncthreads();

    // ---- scatter-add m_aggr (segment_sum over col)
    {
        int e = tid >> 1, kb = (tid & 1) * 64;
        int cnode = eidx[NE + e0 + e];
        float* dst = maggr + (long)cnode * 128 + kb;
        for (int k = 0; k < 64; ++k) atomicAdd(dst + k, (float)sm_m[e * 128 + kb + k]);
    }

    // ---- coord weights: c3 = silu(m @ Wc1 + bc1); cw = c3 @ Wc2 + bc2
    const v16h* B3 = (const v16h*)(sm_bw + 16384);   // Wc1 frags (LDS)
#pragma unroll
    for (int nt = 0; nt < 8; ++nt) {
        float b = bc1[nt * 16 + m15];
        v8f c;
#pragma unroll
        for (int i = 0; i < 8; ++i) c[i] = b;
#pragma unroll
        for (int kt = 0; kt < 4; ++kt) {
            v16h a = load_a_frag(sm_m, 128, row_base, kt * 32);
            c = wmma_f16(a, B3[(kt * 8 + nt) * 32 + lane], c);
        }
        acc[nt] = c;
    }
    // reuse sm_msg (free since layer1) as c3 buffer, stride 128
#pragma unroll
    for (int nt = 0; nt < 8; ++nt)
#pragma unroll
        for (int v = 0; v < 8; ++v)
            sm_msg[(row_base + v + 8 * hi) * 128 + nt * 16 + m15] = (_Float16)siluf_(acc[nt][v]);
    __syncthreads();

    if (tid < 64) {
        float s = bc2[0];
        const v8h* row = (const v8h*)&sm_msg[tid * 128];
        for (int k8 = 0; k8 < 16; ++k8) {
            v8h v = row[k8];
#pragma unroll
            for (int i = 0; i < 8; ++i) s += (float)v[i] * Wc2[k8 * 8 + i];
        }
        int cnode = eidx[NE + e0 + tid];
        atomicAdd(&xout[cnode * 3 + 0], sm_cd[tid * 3 + 0] * s);
        atomicAdd(&xout[cnode * 3 + 1], sm_cd[tid * 3 + 1] * s);
        atomicAdd(&xout[cnode * 3 + 2], sm_cd[tid * 3 + 2] * s);
    }
}

// ---------------------------------------------------------------- node kernel
__global__ __launch_bounds__(128)
void egnn_node_kernel(const float* __restrict__ h, const float* __restrict__ maggr,
                      const _Float16* __restrict__ Wn1f, const float* __restrict__ bn1,
                      const _Float16* __restrict__ Wn2f, const float* __restrict__ bn2,
                      float* __restrict__ hout) {
    __shared__ _Float16 sm_in[64 * 256];    // [64 nodes][h | m_aggr]
    __shared__ _Float16 sm_l1[64 * 128];
    __shared__ __align__(64) _Float16 sm_b2[16384];   // TDM-staged Wn2 frags (32KB)

    const int tid  = threadIdx.x;
    const int wave = tid >> 5;
    const int lane = tid & 31;
    const int n0   = blockIdx.x * 64;

    if (wave == 0)
        tdm_load_row((unsigned)(unsigned long long)&sm_b2[0], Wn2f, 4096 /* x8B = 32KB */);

    for (int idx = wave; idx < 128; idx += 4) {
        int e = idx >> 1, part = idx & 1;
        int node = n0 + e;
        if (node >= NN) node = NN - 1;   // clamp loads; stores guarded below
        const float* src = part ? (maggr + (long)node * 128) : (h + (long)node * 128);
        float4 v = ((const float4*)src)[lane];
        v4h p;
        p[0] = (_Float16)v.x; p[1] = (_Float16)v.y;
        p[2] = (_Float16)v.z; p[3] = (_Float16)v.w;
        *(v4h*)&sm_in[e * 256 + part * 128 + lane * 4] = p;
    }
    __syncthreads();

    const int m15 = lane & 15, hi = lane >> 4;
    const int row_base = wave * 16;
    v8f acc[8];

    // layer 1: [64x256] @ [256x128], B from global
    const v16h* B1 = (const v16h*)Wn1f;
#pragma unroll
    for (int nt = 0; nt < 8; ++nt) {
        float b = bn1[nt * 16 + m15];
        v8f c;
#pragma unroll
        for (int i = 0; i < 8; ++i) c[i] = b;
#pragma unroll
        for (int kt = 0; kt < 8; ++kt) {
            v16h a = load_a_frag(sm_in, 256, row_base, kt * 32);
            c = wmma_f16(a, B1[(kt * 8 + nt) * 32 + lane], c);
        }
        acc[nt] = c;
    }
#pragma unroll
    for (int nt = 0; nt < 8; ++nt)
#pragma unroll
        for (int v = 0; v < 8; ++v)
            sm_l1[(row_base + v + 8 * hi) * 128 + nt * 16 + m15] = (_Float16)siluf_(acc[nt][v]);

    if (wave == 0) __builtin_amdgcn_s_wait_tensorcnt(0);
    __syncthreads();

    // layer 2 + residual: hout = h + (l1 @ Wn2 + bn2), B from LDS (TDM-staged)
    const v16h* B2 = (const v16h*)sm_b2;
#pragma unroll
    for (int nt = 0; nt < 8; ++nt) {
        float b = bn2[nt * 16 + m15];
        v8f c;
#pragma unroll
        for (int i = 0; i < 8; ++i) c[i] = b;
#pragma unroll
        for (int kt = 0; kt < 4; ++kt) {
            v16h a = load_a_frag(sm_l1, 128, row_base, kt * 32);
            c = wmma_f16(a, B2[(kt * 8 + nt) * 32 + lane], c);
        }
#pragma unroll
        for (int v = 0; v < 8; ++v) {
            int node = n0 + row_base + v + 8 * hi;
            if (node < NN) {
                int col = nt * 16 + m15;
                hout[(long)node * 128 + col] = h[(long)node * 128 + col] + c[v];
            }
        }
    }
}

// ---------------------------------------------------------------- launcher
extern "C" void kernel_launch(void* const* d_in, const int* in_sizes, int n_in,
                              void* d_out, int out_size, void* d_ws, size_t ws_size,
                              hipStream_t stream) {
    (void)in_sizes; (void)n_in; (void)out_size; (void)ws_size;

    const float* h    = (const float*)d_in[0];
    const float* x    = (const float*)d_in[1];
    const int*   eidx = (const int*)  d_in[2];
    const float* We1  = (const float*)d_in[3];
    const float* be1  = (const float*)d_in[4];
    const float* We2  = (const float*)d_in[5];
    const float* be2  = (const float*)d_in[6];
    const float* Wg   = (const float*)d_in[7];
    const float* bg   = (const float*)d_in[8];
    const float* Wn1  = (const float*)d_in[9];
    const float* bn1  = (const float*)d_in[10];
    const float* Wn2  = (const float*)d_in[11];
    const float* bn2  = (const float*)d_in[12];
    const float* Wc1  = (const float*)d_in[13];
    const float* bc1  = (const float*)d_in[14];
    const float* Wc2  = (const float*)d_in[15];
    const float* bc2  = (const float*)d_in[16];

    char* ws = (char*)d_ws;
    float*    maggr = (float*)   (ws + 0);              // NN*128 f32 = 25,600,000 B
    _Float16* We1f  = (_Float16*)(ws + 25600000);       // 9*8*32*16 halves = 73,728 B
    _Float16* We2f  = (_Float16*)(ws + 25673728);       // 32,768 B   (contiguous with Wc1f:
    _Float16* Wc1f  = (_Float16*)(ws + 25706496);       // 32,768 B    one 64KB TDM stage)
    _Float16* Wn1f  = (_Float16*)(ws + 25739264);       // 65,536 B
    _Float16* Wn2f  = (_Float16*)(ws + 25804800);       // 32,768 B

    float* hout = (float*)d_out;
    float* xout = hout + (long)NN * 128;

    init_kernel<<<512, 256, 0, stream>>>(maggr, x, xout);
    pack_b_kernel<<<9, 256, 0, stream>>>(We1, We1f, 257, 9);
    pack_b_kernel<<<4, 256, 0, stream>>>(We2, We2f, 128, 4);
    pack_b_kernel<<<4, 256, 0, stream>>>(Wc1, Wc1f, 128, 4);
    pack_b_kernel<<<8, 256, 0, stream>>>(Wn1, Wn1f, 256, 8);
    pack_b_kernel<<<4, 256, 0, stream>>>(Wn2, Wn2f, 128, 4);

    egnn_edge_kernel<<<NE / 64, 128, 0, stream>>>(h, x, eidx,
                                                  We1f, be1, We2f /* + Wc1f */, be2,
                                                  Wg, bg, bc1, Wc2, bc2,
                                                  maggr, xout);
    egnn_node_kernel<<<(NN + 63) / 64, 128, 0, stream>>>(h, maggr, Wn1f, bn1, Wn2f, bn2, hout);
}
